// DeformableConvBlock_1159641170553
// MI455X (gfx1250) — matrix operand order repaired and checked by reading, not compiled, not run on hardware
//
#include <hip/hip_runtime.h>
#include <hip/hip_bf16.h>

typedef __attribute__((ext_vector_type(16))) __bf16 v16bf;
typedef __attribute__((ext_vector_type(8)))  float  v8f;

#define BATCH 4
#define CIN   256
#define COUT  256
#define HH    64
#define WW    64
#define HW    (HH*WW)          // 4096
#define KTAPS 9
#define KDIM  (CIN*KTAPS)      // 2304
#define NSTEPS (KDIM/32)       // 72

struct alignas(16) CoordEnt {
    int   i00, i01, i10, i11;      // clamped flat spatial indices
    float w00, w01, w10, w11;      // bilinear weights, zeroed when OOB
};

static __device__ __forceinline__ unsigned short f2bf(float f) {
    unsigned int u = __float_as_uint(f);
    unsigned int r = u + 0x7FFFu + ((u >> 16) & 1u);   // round-to-nearest-even
    return (unsigned short)(r >> 16);
}

// ---------------------------------------------------------------------------
// Kernel 1: 18-channel offset conv + bilinear coordinate/weight precompute.
// One thread per (b, pos). Weight addresses are lane-uniform -> scalar loads.
// ---------------------------------------------------------------------------
__global__ __launch_bounds__(256) void offs_coords_kernel(
    const float* __restrict__ x, const float* __restrict__ ow,
    const float* __restrict__ ob, CoordEnt* __restrict__ coords)
{
    const int b   = blockIdx.x >> 4;
    const int pos = ((blockIdx.x & 15) << 8) + threadIdx.x;
    const int ho  = pos >> 6;
    const int wo  = pos & 63;

    float acc[18];
#pragma unroll
    for (int ch = 0; ch < 18; ++ch) acc[ch] = ob[ch];

    const float* xb = x + (size_t)b * CIN * HW;
    for (int c = 0; c < CIN; ++c) {
        const float* xc = xb + c * HW;
        float xv[9];
#pragma unroll
        for (int t = 0; t < 9; ++t) {
            int yy = ho - 1 + t / 3;
            int xx = wo - 1 + t % 3;
            bool ok = (yy >= 0) & (yy < HH) & (xx >= 0) & (xx < WW);
            xv[t] = ok ? xc[yy * WW + xx] : 0.0f;
        }
        const float* wc = ow + c * 9;   // offset_w[ch][c][t] = ow[ch*2304 + c*9 + t]
#pragma unroll
        for (int ch = 0; ch < 18; ++ch) {
            const float* w = wc + ch * KDIM;
#pragma unroll
            for (int t = 0; t < 9; ++t) acc[ch] = fmaf(xv[t], w[t], acc[ch]);
        }
    }

#pragma unroll
    for (int k = 0; k < 9; ++k) {
        float py = (float)(ho - 1 + k / 3) + acc[2 * k];
        float px = (float)(wo - 1 + k % 3) + acc[2 * k + 1];
        float fy = floorf(py), fx = floorf(px);
        int y0 = (int)fy, x0 = (int)fx;
        int y1 = y0 + 1,  x1 = x0 + 1;
        float wy = py - fy, wx = px - fx;
        bool vy0 = (y0 >= 0) & (y0 < HH), vy1 = (y1 >= 0) & (y1 < HH);
        bool vx0 = (x0 >= 0) & (x0 < WW), vx1 = (x1 >= 0) & (x1 < WW);
        int y0c = min(max(y0, 0), HH - 1), y1c = min(max(y1, 0), HH - 1);
        int x0c = min(max(x0, 0), WW - 1), x1c = min(max(x1, 0), WW - 1);
        CoordEnt e;
        e.i00 = y0c * WW + x0c;  e.i01 = y0c * WW + x1c;
        e.i10 = y1c * WW + x0c;  e.i11 = y1c * WW + x1c;
        e.w00 = (vy0 & vx0) ? (1.f - wy) * (1.f - wx) : 0.f;
        e.w01 = (vy0 & vx1) ? (1.f - wy) * wx         : 0.f;
        e.w10 = (vy1 & vx0) ? wy * (1.f - wx)         : 0.f;
        e.w11 = (vy1 & vx1) ? wy * wx                 : 0.f;
        coords[(((b * 9) + k) << 12) + pos] = e;
    }
}

// ---------------------------------------------------------------------------
// Kernel 2: pack dcn_w (f32 [COUT][CIN][9]) into bf16 WMMA A-fragment order.
// GEMM K-dim ordering: kd = tap*256 + c.  Per (mblk, kstep): 32 lanes x 16
// halves, lane holds M = mblk*16 + (lane&15); element e -> K per ISA layout.
// ---------------------------------------------------------------------------
__global__ __launch_bounds__(256) void apack_kernel(
    const float* __restrict__ wk, unsigned short* __restrict__ apack)
{
    const int tid  = blockIdx.x * 256 + threadIdx.x;    // 16*72*32 = 36864
    const int lane = tid & 31;
    const int s    = (tid >> 5) % NSTEPS;
    const int mblk = tid / (32 * NSTEPS);
    const int m    = (mblk << 4) + (lane & 15);
    const int kadd = (lane & 16) ? 8 : 0;

    unsigned int d[8];
#pragma unroll
    for (int i = 0; i < 8; ++i) {
        unsigned short v2[2];
#pragma unroll
        for (int h = 0; h < 2; ++h) {
            int e  = 2 * i + h;
            int kl = ((e < 8) ? e : e + 8) + kadd;      // ISA 16-bit A layout
            int kd = (s << 5) + kl;
            int kt = kd >> 8;                           // tap index
            int c  = kd & 255;                          // input channel
            v2[h] = f2bf(wk[(m * CIN + c) * 9 + kt]);
        }
        d[i] = (unsigned int)v2[0] | ((unsigned int)v2[1] << 16);
    }
    uint4* dst = (uint4*)(apack + (size_t)tid * 16);
    dst[0] = make_uint4(d[0], d[1], d[2], d[3]);
    dst[1] = make_uint4(d[4], d[5], d[6], d[7]);
}

// ---------------------------------------------------------------------------
// Kernel 3: fused bilinear-gather + bf16 WMMA GEMM.
// Grid (64, 2, B); block 256 = 8 waves. WG computes C tile [128 x 64].
// Per K-step: all threads gather a 32(c) x 64(pos) bf16 tile into LDS
// (double buffered, row stride 40 halves to avoid bank conflicts), then each
// wave does 4x v_wmma_f32_16x16x32_bf16.
// ---------------------------------------------------------------------------
__global__ __launch_bounds__(256) void dcn_gemm_kernel(
    const float* __restrict__ x, const CoordEnt* __restrict__ coords,
    const unsigned short* __restrict__ apack, const float* __restrict__ bias,
    float* __restrict__ out)
{
    __shared__ alignas(16) unsigned short tile[2][64 * 40];

    const int tid   = threadIdx.x;
    const int lane  = tid & 31;
    const int wave  = tid >> 5;
    const int nblk  = blockIdx.x;      // 64 position blocks of 64
    const int mhalf = blockIdx.y;      // 2 halves of COUT
    const int b     = blockIdx.z;
    const int posg0 = nblk << 6;

    // gather mapping: thread -> one position, 8 channels
    const int gpos = tid & 63;
    const int gc8  = (tid >> 6) << 3;  // 0,8,16,24
    const int posg = posg0 + gpos;

    const int    mblk  = (mhalf << 3) + wave;
    const size_t abase = (size_t)mblk * NSTEPS * 32 * 16 + (size_t)lane * 16;

    const float* xb = x + (size_t)b * CIN * HW;

    v8f acc0 = {}, acc1 = {}, acc2 = {}, acc3 = {};

    auto gather = [&](int s, int buf) {
        const int kidx = s >> 3;              // tap constant within a step
        const int c0   = (s & 7) << 5;
        CoordEnt e = coords[(((b * 9) + kidx) << 12) + posg];
        const float* xc = xb + (size_t)(c0 + gc8) * HW;
        unsigned int d[4];
#pragma unroll
        for (int i = 0; i < 4; ++i) {
            float v0 = e.w00 * xc[e.i00] + e.w01 * xc[e.i01]
                     + e.w10 * xc[e.i10] + e.w11 * xc[e.i11];
            xc += HW;
            float v1 = e.w00 * xc[e.i00] + e.w01 * xc[e.i01]
                     + e.w10 * xc[e.i10] + e.w11 * xc[e.i11];
            xc += HW;
            d[i] = (unsigned int)f2bf(v0) | ((unsigned int)f2bf(v1) << 16);
        }
        *(uint4*)&tile[buf][gpos * 40 + gc8] = make_uint4(d[0], d[1], d[2], d[3]);
    };

    union ABFrag { uint4 q[2]; v16bf v; };

    gather(0, 0);
    __syncthreads();

    for (int s = 0; s < NSTEPS; ++s) {
        const int buf = s & 1;
        if (s + 1 < NSTEPS) gather(s + 1, buf ^ 1);

        ABFrag a;
        const uint4* ap = (const uint4*)(apack + abase + (size_t)s * 512);
        a.q[0] = ap[0];
        a.q[1] = ap[1];

        const int koff = (lane & 16) >> 1;  // B layout: lanes>=16 hold K=8..15,24..31
#pragma unroll
        for (int nt = 0; nt < 4; ++nt) {
            ABFrag bm;
            const uint4* bp =
                (const uint4*)&tile[buf][(nt * 16 + (lane & 15)) * 40 + koff];
            bm.q[0] = bp[0];
            bm.q[1] = bp[2];
            v8f& acc = nt == 0 ? acc0 : (nt == 1 ? acc1 : (nt == 2 ? acc2 : acc3));
            acc = __builtin_amdgcn_wmma_f32_16x16x32_bf16(
                false, a.v, false, bm.v, (short)0, acc, false, false);
        }
        __syncthreads();
    }

    // Epilogue: C layout -> lane n = lane&15, row = r + 8*(lane>=16)
    const int mrow = (mhalf << 7) + (wave << 4);
    const int rsel = (lane >> 4) << 3;
    const int n    = lane & 15;
#pragma unroll
    for (int nt = 0; nt < 4; ++nt) {
        v8f acc = nt == 0 ? acc0 : (nt == 1 ? acc1 : (nt == 2 ? acc2 : acc3));
        const int pos = posg0 + nt * 16 + n;
#pragma unroll
        for (int r = 0; r < 8; ++r) {
            const int o = mrow + rsel + r;
            out[(((size_t)b * COUT + o) << 12) + pos] = acc[r] + bias[o];
        }
    }
}

// ---------------------------------------------------------------------------
extern "C" void kernel_launch(void* const* d_in, const int* in_sizes, int n_in,
                              void* d_out, int out_size, void* d_ws, size_t ws_size,
                              hipStream_t stream)
{
    const float* x        = (const float*)d_in[0];   // [B,CIN,H,W]
    const float* offset_w = (const float*)d_in[1];   // [18,CIN,3,3]
    const float* offset_b = (const float*)d_in[2];   // [18]
    const float* dcn_w    = (const float*)d_in[3];   // [COUT,CIN,3,3]
    const float* dcn_b    = (const float*)d_in[4];   // [COUT]
    float*       out      = (float*)d_out;           // [B,COUT,H,W]

    CoordEnt* coords = (CoordEnt*)d_ws;                                  // 4.72 MB
    unsigned short* apack =
        (unsigned short*)((char*)d_ws + (size_t)BATCH * 9 * HW * sizeof(CoordEnt)); // 1.18 MB

    offs_coords_kernel<<<64, 256, 0, stream>>>(x, offset_w, offset_b, coords);
    apack_kernel<<<144, 256, 0, stream>>>(dcn_w, apack);
    dcn_gemm_kernel<<<dim3(64, 2, BATCH), 256, 0, stream>>>(x, coords, apack, dcn_b, out);
}
